// Model_85160611545254
// MI455X (gfx1250) — matrix operand order repaired and checked by reading, not compiled, not run on hardware
//
#include <hip/hip_runtime.h>
#include <hip/hip_bf16.h>
#include <cstdint>

#define LNUM   4
#define DMODEL 512
#define NHEAD  8
#define HDIM   64
#define DFFN   2048
#define BBATCH 8
#define SLEN   1024
#define NTOK   (BBATCH*SLEN)

typedef __attribute__((ext_vector_type(16))) __bf16 v16bf;
typedef __attribute__((ext_vector_type(8)))  float  v8f;

__device__ inline unsigned short f2bf(float f) {
  unsigned int u = __float_as_uint(f);
  return (unsigned short)((u + 0x7FFFu + ((u >> 16) & 1u)) >> 16);  // RNE
}

// A-operand fragment: 16x32 bf16, row = lane&15.
// Per ISA table: lanes 0-15 hold K={0..7, 16..23}, lanes 16-31 K={8..15, 24..31}.
__device__ inline v16bf frag_ld_a(const unsigned short* p, int half) {
  union { v16bf v; uint4 u[2]; } r;
  r.u[0] = *(const uint4*)(p + half * 8);
  r.u[1] = *(const uint4*)(p + 16 + half * 8);
  return r.v;
}

// B-operand fragment: 32x16 bf16 from a [N][K] (pre-transposed) matrix,
// column n = lane&15; lanes 0-15 hold K={0..7, 16..23}, lanes 16-31 K={8..15, 24..31}.
__device__ inline v16bf frag_ld_b(const unsigned short* p, int half) {
  union { v16bf v; uint4 u[2]; } r;
  r.u[0] = *(const uint4*)(p + half * 8);
  r.u[1] = *(const uint4*)(p + 16 + half * 8);
  return r.v;
}

__device__ inline v8f wmma_bf16(v16bf a, v16bf b, v8f c) {
  return __builtin_amdgcn_wmma_f32_16x16x32_bf16(false, a, false, b,
                                                 (short)0, c, false, false);
}

// ---- 16-lane xor-butterfly reductions via DPP ROW_XMASK (no LDS traffic) ----
template <int XM>
__device__ inline float dpp_xor16(float x) {
  int v = __builtin_amdgcn_update_dpp(0, __float_as_int(x), 0x160 | XM, 0xf, 0xf,
                                      true);
  return __int_as_float(v);
}
__device__ inline float red_max16(float x) {
  x = fmaxf(x, dpp_xor16<1>(x));
  x = fmaxf(x, dpp_xor16<2>(x));
  x = fmaxf(x, dpp_xor16<4>(x));
  x = fmaxf(x, dpp_xor16<8>(x));
  return x;
}
__device__ inline float red_sum16(float x) {
  x += dpp_xor16<1>(x);
  x += dpp_xor16<2>(x);
  x += dpp_xor16<4>(x);
  x += dpp_xor16<8>(x);
  return x;
}

// ---- CDNA5 async global->LDS copy (ASYNCcnt-tracked, no VGPR staging) -------
__device__ inline void async_ld_b128(unsigned lds_off, const void* gaddr) {
  asm volatile("global_load_async_to_lds_b128 %0, %1, off"
               :: "v"(lds_off), "v"(gaddr) : "memory");
}
__device__ inline void wait_async_le2() {
  asm volatile("s_wait_asynccnt 0x2" ::: "memory");
}
__device__ inline void wait_async_0() {
  asm volatile("s_wait_asynccnt 0x0" ::: "memory");
}

// ---------------- weight convert + transpose: W[K][N] f32 -> Wt[N][K] bf16 ----
__global__ void wconv(const float* __restrict__ W, unsigned short* __restrict__ Wt,
                      int Kd, int Nd) {
  int idx = blockIdx.x * 256 + threadIdx.x;
  if (idx >= Kd * Nd) return;
  int k = idx / Nd, n = idx % Nd;
  Wt[(size_t)n * Kd + k] = f2bf(W[idx]);
}

// ---------------- x + positional encoding -> fp32 residual stream ------------
__global__ void addpos(const float* __restrict__ X, float* __restrict__ Out) {
  int idx = blockIdx.x * 256 + threadIdx.x;
  int c = idx & (DMODEL - 1);
  int s = (idx / DMODEL) & (SLEN - 1);
  float p = (float)s;
  const float LN1E4 = 9.210340371976184f;
  float pe;
  if (c < DMODEL / 2) {
    pe = __sinf(p * __expf(-((float)c / (DMODEL / 2)) * LN1E4));
  } else {
    pe = __cosf(p * __expf(-((float)(c - DMODEL / 2) / (DMODEL / 2)) * LN1E4));
  }
  Out[idx] = X[idx] + pe;
}

// ---------------- layernorm (fp32 in, bf16 out), one wave per token ----------
__global__ __launch_bounds__(256) void ln_bf16(const float* __restrict__ X,
                                               const float* __restrict__ G,
                                               const float* __restrict__ Bv,
                                               unsigned short* __restrict__ Out) {
  int token = blockIdx.x * 8 + (threadIdx.x >> 5);
  int lane = threadIdx.x & 31;
  const float* xr = X + (size_t)token * DMODEL;
  float v[16];
  float s = 0.f;
#pragma unroll
  for (int i = 0; i < 16; ++i) { v[i] = xr[lane + i * 32]; s += v[i]; }
  s = red_sum16(s); s += __shfl_xor(s, 16, 32);
  float mu = s * (1.f / DMODEL);
  float var = 0.f;
#pragma unroll
  for (int i = 0; i < 16; ++i) { float d = v[i] - mu; var += d * d; }
  var = red_sum16(var); var += __shfl_xor(var, 16, 32);
  float inv = rsqrtf(var * (1.f / DMODEL) + 1e-3f);
  unsigned short* orow = Out + (size_t)token * DMODEL;
#pragma unroll
  for (int i = 0; i < 16; ++i) {
    int c = lane + i * 32;
    orow[c] = f2bf((v[i] - mu) * inv * G[c] + Bv[c]);
  }
}

// ---------------- WMMA GEMM: C[M][N] = A[M][K](bf16) x Bt[N][K](bf16) --------
// Block: 256 thr (8 waves), tile 256(M) x 64(N); each wave owns a 32x64 tile.
// B tile (shared across waves) is staged in LDS via double-buffered
// GLOBAL_LOAD_ASYNC_TO_LDS_B128 (64-wide K chunks, 2 async instrs/wave/chunk;
// s_wait_asynccnt 2 retires the in-flight chunk since async loads complete
// in order per wave). A fragments (wave-private rows) load direct from global.
// MODE 0: store bf16 [M][N]         MODE 1: store bf16 V-transposed [B,H,HD,S]
// MODE 2: Res[M][N] += C (+bias)    MODE 3: relu(C + bias) -> bf16 [M][N]
template <int MODE>
__global__ __launch_bounds__(256) void gemm_bf16(
    const unsigned short* __restrict__ A, const unsigned short* __restrict__ Bt,
    const float* __restrict__ bias, unsigned short* __restrict__ OutB,
    float* __restrict__ Res, int M, int N, int K) {
  __shared__ unsigned short btile[2][64][64];  // [buf][n][k], 16 KB
  int tid = threadIdx.x;
  int wave = tid >> 5;
  int lane = tid & 31;
  int lr = lane & 15, half = lane >> 4;
  int rowBase = blockIdx.y * 256 + wave * 32;
  int colBase = blockIdx.x * 64;
  const unsigned short* Bbase = Bt + (size_t)colBase * K;

  auto issue = [&](int kc, int buf) {
    // 64 rows x 128B = 512 b128 transfers; 2 per thread
#pragma unroll
    for (int r = 0; r < 2; ++r) {
      int i = tid + r * 256;
      int n = i >> 3, sg = i & 7;
      const unsigned short* gp = Bbase + (size_t)n * K + kc + sg * 8;
      unsigned loff = (unsigned)(uintptr_t)&btile[buf][n][sg * 8];
      async_ld_b128(loff, gp);
    }
  };

  v8f z = {};
  v8f acc[8] = {z, z, z, z, z, z, z, z};
  const unsigned short* Arow0 = A + (size_t)(rowBase + lr) * K;
  const unsigned short* Arow1 = A + (size_t)(rowBase + 16 + lr) * K;
  int nchunk = K >> 6;
  issue(0, 0);
  for (int c = 0; c < nchunk; ++c) {
    int buf = c & 1;
    if (c + 1 < nchunk) {
      issue((c + 1) << 6, buf ^ 1);
      wait_async_le2();   // chunk c done (per-wave in-order completion)
    } else {
      wait_async_0();
    }
    __syncthreads();      // chunk c visible from every wave
#pragma unroll
    for (int kk = 0; kk < 2; ++kk) {
      int k0 = (c << 6) + kk * 32;
      v16bf a0 = frag_ld_a(Arow0 + k0, half);
      v16bf a1 = frag_ld_a(Arow1 + k0, half);
#pragma unroll
      for (int g = 0; g < 4; ++g) {
        v16bf b = frag_ld_b(&btile[buf][g * 16 + lr][kk * 32], half);
        acc[g]     = wmma_bf16(a0, b, acc[g]);
        acc[4 + g] = wmma_bf16(a1, b, acc[4 + g]);
      }
    }
    __syncthreads();      // all waves done reading buf before it is re-filled
  }

#pragma unroll
  for (int t = 0; t < 2; ++t) {
#pragma unroll
    for (int g = 0; g < 4; ++g) {
      int n = colBase + g * 16 + lr;
      float bv = bias ? bias[n] : 0.f;
#pragma unroll
      for (int j = 0; j < 8; ++j) {
        int m = rowBase + t * 16 + half * 8 + j;  // C layout: VGPR j -> row half*8+j
        float c = acc[t * 4 + g][j] + bv;
        if (MODE == 0) {
          OutB[(size_t)m * N + n] = f2bf(c);
        } else if (MODE == 1) {
          int bb = m / SLEN, ss = m % SLEN;
          int hh = n / HDIM, hd = n % HDIM;
          OutB[(((size_t)bb * NHEAD + hh) * HDIM + hd) * SLEN + ss] = f2bf(c);
        } else if (MODE == 2) {
          Res[(size_t)m * N + n] += c;
        } else {
          OutB[(size_t)m * N + n] = f2bf(fmaxf(c, 0.f));
        }
      }
    }
  }
}

// ---------------- fused flash attention ---------------------------------------
// grid (S/128, B*H); 8 waves/block, each wave owns a 16-query tile and loops
// over keys 64 at a time (8 score WMMAs + 8 PV WMMAs per step).
// Q,K: bf16 [B*S][D] (head cols h*64..); Vt: bf16 [B,H,HD,S]; O: bf16 [B*S][D].
__global__ __launch_bounds__(256) void attn_fused(
    const unsigned short* __restrict__ Q, const unsigned short* __restrict__ Kc,
    const unsigned short* __restrict__ Vt, const unsigned char* __restrict__ mask,
    unsigned short* __restrict__ O) {
  __shared__ unsigned short pls[8][16 * 64];  // per-wave P staging (2 KB each)
  int wave = threadIdx.x >> 5;
  int lane = threadIdx.x & 31;
  int lr = lane & 15, half = lane >> 4;
  int b = blockIdx.y / NHEAD, h = blockIdx.y % NHEAD;
  int q0 = blockIdx.x * 128 + wave * 16;
  const unsigned short* Qrow = Q + (size_t)(b * SLEN + q0 + lr) * DMODEL + h * HDIM;
  v16bf aq0 = frag_ld_a(Qrow, half);        // contraction 0..31
  v16bf aq1 = frag_ld_a(Qrow + 32, half);   // contraction 32..63
  const unsigned short* Kbase = Kc + (size_t)b * SLEN * DMODEL + h * HDIM;
  const unsigned short* Vbase = Vt + ((size_t)b * NHEAD + h) * HDIM * SLEN;
  const unsigned char* mrow = mask + b * SLEN;
  float m_i[8], l_i[8];
  v8f z = {};
  v8f o[4] = {z, z, z, z};
#pragma unroll
  for (int j = 0; j < 8; ++j) { m_i[j] = -1e30f; l_i[j] = 0.f; }
  const float scale = 0.125f;  // 1/sqrt(64)

  for (int kk = 0; kk < SLEN; kk += 64) {
    v8f s[4] = {z, z, z, z};
#pragma unroll
    for (int g = 0; g < 4; ++g) {     // K rows [key][hd] == Bt[N=key][K=hd]
      const unsigned short* Kr = Kbase + (size_t)(kk + g * 16 + lr) * DMODEL;
      s[g] = wmma_bf16(aq0, frag_ld_b(Kr, half), s[g]);
      s[g] = wmma_bf16(aq1, frag_ld_b(Kr + 32, half), s[g]);
    }
    float pen[4];  // key mask (assumes 1-byte bools, numpy layout)
#pragma unroll
    for (int g = 0; g < 4; ++g)
      pen[g] = mrow[kk + g * 16 + lr] ? 0.f : -1e9f;
#pragma unroll
    for (int j = 0; j < 8; ++j) {
      float a0 = s[0][j] * scale + pen[0];
      float a1 = s[1][j] * scale + pen[1];
      float a2 = s[2][j] * scale + pen[2];
      float a3 = s[3][j] * scale + pen[3];
      float rmax = red_max16(fmaxf(fmaxf(a0, a1), fmaxf(a2, a3)));
      float mnew = fmaxf(m_i[j], rmax);
      float corr = __expf(m_i[j] - mnew);
      float p0 = __expf(a0 - mnew), p1 = __expf(a1 - mnew);
      float p2 = __expf(a2 - mnew), p3 = __expf(a3 - mnew);
      float rsum = red_sum16((p0 + p1) + (p2 + p3));
      l_i[j] = l_i[j] * corr + rsum;
      m_i[j] = mnew;
      o[0][j] *= corr; o[1][j] *= corr; o[2][j] *= corr; o[3][j] *= corr;
      unsigned short* pr = &pls[wave][(half * 8 + j) * 64];
      pr[lr] = f2bf(p0); pr[16 + lr] = f2bf(p1);
      pr[32 + lr] = f2bf(p2); pr[48 + lr] = f2bf(p3);
    }
    // re-layout P (C-layout) -> A-fragments via LDS (same-wave, no barrier)
    v16bf ap0 = frag_ld_a(&pls[wave][lr * 64], half);
    v16bf ap1 = frag_ld_a(&pls[wave][lr * 64 + 32], half);
#pragma unroll
    for (int g = 0; g < 4; ++g) {    // Vt rows are Bt[N=hd][K=key]
      const unsigned short* Vrow = Vbase + (size_t)(g * 16 + lr) * SLEN + kk;
      o[g] = wmma_bf16(ap0, frag_ld_b(Vrow, half), o[g]);
      o[g] = wmma_bf16(ap1, frag_ld_b(Vrow + 32, half), o[g]);
    }
  }
#pragma unroll
  for (int g = 0; g < 4; ++g) {
#pragma unroll
    for (int j = 0; j < 8; ++j) {
      float val = o[g][j] / l_i[j];
      O[(size_t)(b * SLEN + q0 + half * 8 + j) * DMODEL + h * HDIM + g * 16 + lr] =
          f2bf(val);
    }
  }
}

extern "C" void kernel_launch(void* const* d_in, const int* in_sizes, int n_in,
                              void* d_out, int out_size, void* d_ws, size_t ws_size,
                              hipStream_t stream) {
  const float* x = (const float*)d_in[0];
  const unsigned char* mask = (const unsigned char*)d_in[1];  // bool mask [B][S]
  const float* Wq = (const float*)d_in[2];
  const float* Wk = (const float*)d_in[3];
  const float* Wv = (const float*)d_in[4];
  const float* Wo = (const float*)d_in[5];
  const float* ln1g = (const float*)d_in[6];
  const float* ln1b = (const float*)d_in[7];
  const float* ln2g = (const float*)d_in[8];
  const float* ln2b = (const float*)d_in[9];
  const float* W1 = (const float*)d_in[10];
  const float* b1 = (const float*)d_in[11];
  const float* W2 = (const float*)d_in[12];
  const float* b2 = (const float*)d_in[13];
  float* out = (float*)d_out;  // fp32 residual stream == final output

  // workspace carve-up (bf16 buffers), ~96 MB total
  unsigned short* p = (unsigned short*)d_ws;
  unsigned short* Wqt = p; p += (size_t)LNUM * DMODEL * DMODEL;
  unsigned short* Wkt = p; p += (size_t)LNUM * DMODEL * DMODEL;
  unsigned short* Wvt = p; p += (size_t)LNUM * DMODEL * DMODEL;
  unsigned short* Wot = p; p += (size_t)LNUM * DMODEL * DMODEL;
  unsigned short* W1t = p; p += (size_t)LNUM * DFFN * DMODEL;
  unsigned short* W2t = p; p += (size_t)LNUM * DMODEL * DFFN;
  unsigned short* hbf = p; p += (size_t)NTOK * DMODEL;
  unsigned short* qb  = p; p += (size_t)NTOK * DMODEL;
  unsigned short* kb  = p; p += (size_t)NTOK * DMODEL;
  unsigned short* vtb = p; p += (size_t)NTOK * DMODEL;
  unsigned short* abf = p; p += (size_t)NTOK * DMODEL;
  unsigned short* fbf = p; p += (size_t)NTOK * DFFN;

  const int nDD = DMODEL * DMODEL;
  const int nDF = DMODEL * DFFN;
  for (int l = 0; l < LNUM; ++l) {
    wconv<<<(nDD + 255) / 256, 256, 0, stream>>>(Wq + (size_t)l * nDD, Wqt + (size_t)l * nDD, DMODEL, DMODEL);
    wconv<<<(nDD + 255) / 256, 256, 0, stream>>>(Wk + (size_t)l * nDD, Wkt + (size_t)l * nDD, DMODEL, DMODEL);
    wconv<<<(nDD + 255) / 256, 256, 0, stream>>>(Wv + (size_t)l * nDD, Wvt + (size_t)l * nDD, DMODEL, DMODEL);
    wconv<<<(nDD + 255) / 256, 256, 0, stream>>>(Wo + (size_t)l * nDD, Wot + (size_t)l * nDD, DMODEL, DMODEL);
    wconv<<<(nDF + 255) / 256, 256, 0, stream>>>(W1 + (size_t)l * nDF, W1t + (size_t)l * nDF, DMODEL, DFFN);
    wconv<<<(nDF + 255) / 256, 256, 0, stream>>>(W2 + (size_t)l * nDF, W2t + (size_t)l * nDF, DFFN, DMODEL);
  }
  addpos<<<(NTOK * DMODEL) / 256, 256, 0, stream>>>(x, out);

  dim3 blk(256);
  dim3 gDD(DMODEL / 64, NTOK / 256);
  dim3 gDF(DFFN / 64, NTOK / 256);
  for (int l = 0; l < LNUM; ++l) {
    ln_bf16<<<NTOK / 8, blk, 0, stream>>>(out, ln1g + l * DMODEL, ln1b + l * DMODEL, hbf);
    gemm_bf16<0><<<gDD, blk, 0, stream>>>(hbf, Wqt + (size_t)l * nDD, nullptr, qb, nullptr, NTOK, DMODEL, DMODEL);
    gemm_bf16<0><<<gDD, blk, 0, stream>>>(hbf, Wkt + (size_t)l * nDD, nullptr, kb, nullptr, NTOK, DMODEL, DMODEL);
    gemm_bf16<1><<<gDD, blk, 0, stream>>>(hbf, Wvt + (size_t)l * nDD, nullptr, vtb, nullptr, NTOK, DMODEL, DMODEL);
    attn_fused<<<dim3(SLEN / 128, BBATCH * NHEAD), blk, 0, stream>>>(qb, kb, vtb, mask, abf);
    gemm_bf16<2><<<gDD, blk, 0, stream>>>(abf, Wot + (size_t)l * nDD, nullptr, nullptr, out, NTOK, DMODEL, DMODEL);
    ln_bf16<<<NTOK / 8, blk, 0, stream>>>(out, ln2g + l * DMODEL, ln2b + l * DMODEL, hbf);
    gemm_bf16<3><<<gDF, blk, 0, stream>>>(hbf, W1t + (size_t)l * nDF, b1 + l * DFFN, fbf, nullptr, NTOK, DFFN, DMODEL);
    gemm_bf16<2><<<gDD, blk, 0, stream>>>(fbf, W2t + (size_t)l * nDF, b2 + l * DMODEL, nullptr, out, NTOK, DMODEL, DFFN);
  }
}
